// MultiHeadAttentionALiBi_48043504173473
// MI455X (gfx1250) — compile-verified, matrix operand order checked by reading, not run on hardware
//
#include <hip/hip_runtime.h>
#include <hip/hip_bf16.h>
#include <math.h>

// ---------------------------------------------------------------------------
// MultiHeadAttention + ALiBi for MI455X (gfx1250, wave32, WMMA).
// B=4, T=1024, C=1024, H=16, hs=64.  Outputs: out[B,T,C] f32 ++ attn[B,H,T,T] f32.
// All matmuls go through v_wmma_f32_16x16x32_f16 (f16 in, f32 accumulate).
// ---------------------------------------------------------------------------

typedef _Float16 half_t;
typedef __attribute__((ext_vector_type(16))) _Float16 v16h;
typedef __attribute__((ext_vector_type(8)))  _Float16 v8h;
typedef __attribute__((ext_vector_type(8)))  float    v8f;

static constexpr int kB = 4, kT = 1024, kC = 1024, kH = 16, kHS = 64;

#define DEVINL __device__ __forceinline__

DEVINL v8f wmma_f16(v16h a, v16h b, v8f c) {
  // D = A(16x32,f16) * B(32x16,f16) + C(16x16,f32)
  return __builtin_amdgcn_wmma_f32_16x16x32_f16(
      /*neg_a=*/false, a, /*neg_b=*/false, b,
      /*c_mod=*/(short)0, c, /*reuse_a=*/false, /*reuse_b=*/false);
}

union V16U { v16h v; v8h h[2]; };

// Load one 16x32 f16 A-fragment (or 32x16 B-fragment, by symmetry) from a
// row-major tile: per ISA layout, lane L (L%16 = row/M or col/N), lane-group
// g=L/16 selects the K interleave; each lane needs K = [8g..8g+7] and
// [16+8g..16+8g+7] -> two contiguous 16-byte loads.
DEVINL v16h load_frag(const half_t* __restrict__ base, int row0, int pitch,
                      int k, int lane) {
  const int lm = lane & 15, lg = lane >> 4;
  const half_t* p = base + (size_t)(row0 + lm) * (size_t)pitch + k + 8 * lg;
  V16U u;
  u.h[0] = *(const v8h*)(p);
  u.h[1] = *(const v8h*)(p + 16);
  return u.v;
}

// ---------------------------------------------------------------------------
// fp32 -> f16 cast
// ---------------------------------------------------------------------------
__global__ __launch_bounds__(256) void mha_cvt_f32_f16(
    const float* __restrict__ in, half_t* __restrict__ out, int n) {
  int i = blockIdx.x * 256 + threadIdx.x;
  if (i < n) out[i] = (half_t)in[i];
}

// ---------------------------------------------------------------------------
// Generic WMMA GEMM:  C[M,N] = A[M,K](f16) * B[N,K](f16)^T + bias[N]
// (B given row-major as [N,K], i.e. torch Linear weight layout.)
// Block = 128x128 (8 waves as 2x4), wave = 64x32 (4x2 D-tiles).
// ---------------------------------------------------------------------------
template <bool OUT_F16>
__global__ __launch_bounds__(256) void mha_gemm_wmma(
    const half_t* __restrict__ A, const half_t* __restrict__ Bm,
    const float* __restrict__ bias, void* __restrict__ Cout,
    int M, int N, int K) {
  const int tid = threadIdx.x, wid = tid >> 5, lane = tid & 31;
  const int lm = lane & 15, lg = lane >> 4;
  const int wm = wid >> 2, wn = wid & 3;
  const int m0 = blockIdx.y * 128 + wm * 64;
  const int n0 = blockIdx.x * 128 + wn * 32;
  (void)M;

  v8f acc[4][2];
#pragma unroll
  for (int mi = 0; mi < 4; ++mi)
#pragma unroll
    for (int ni = 0; ni < 2; ++ni) acc[mi][ni] = (v8f){};

  for (int k = 0; k < K; k += 32) {
    v16h af[4], bf[2];
#pragma unroll
    for (int mi = 0; mi < 4; ++mi) af[mi] = load_frag(A, m0 + mi * 16, K, k, lane);
#pragma unroll
    for (int ni = 0; ni < 2; ++ni) bf[ni] = load_frag(Bm, n0 + ni * 16, K, k, lane);
#pragma unroll
    for (int mi = 0; mi < 4; ++mi)
#pragma unroll
      for (int ni = 0; ni < 2; ++ni)
        acc[mi][ni] = wmma_f16(af[mi], bf[ni], acc[mi][ni]);
  }

  // D layout: VGPR r -> row (r + 8*lg), col = lm  (per tile)
#pragma unroll
  for (int ni = 0; ni < 2; ++ni) {
    const int n = n0 + ni * 16 + lm;
    const float bn = bias[n];
#pragma unroll
    for (int mi = 0; mi < 4; ++mi) {
#pragma unroll
      for (int r = 0; r < 8; ++r) {
        const size_t m = (size_t)(m0 + mi * 16 + r + 8 * lg);
        const float v = acc[mi][ni][r] + bn;
        if (OUT_F16) ((half_t*)Cout)[m * (size_t)N + n] = (half_t)v;
        else         ((float*)Cout)[m * (size_t)N + n] = v;
      }
    }
  }
}

// ---------------------------------------------------------------------------
// Vt[b,h,d,t] = qkv[b,t, 2C + h*64 + d]   (f16, makes PV B-fragments row-major)
// ---------------------------------------------------------------------------
__global__ __launch_bounds__(256) void mha_vtrans(
    const half_t* __restrict__ qkv, half_t* __restrict__ vt) {
  size_t idx = (size_t)blockIdx.x * 256 + threadIdx.x;  // b,h,d,t ; t fastest
  const int t = (int)(idx & (kT - 1));
  size_t r = idx >> 10;
  const int d = (int)(r & (kHS - 1)); r >>= 6;
  const int h = (int)(r & (kH - 1));
  const int b = (int)(r >> 4);
  vt[idx] = qkv[((size_t)b * kT + t) * (3 * kC) + 2 * kC + h * kHS + d];
}

// ---------------------------------------------------------------------------
// Attention: one workgroup per (b, h, 16-row q tile). 8 waves; wave w owns
// columns [w*128, w*128+128). Scores live in registers (8 D-tiles / wave).
// ---------------------------------------------------------------------------
__global__ __launch_bounds__(256) void mha_attn_kernel(
    const half_t* __restrict__ qkv, const half_t* __restrict__ vt,
    float* __restrict__ attn, half_t* __restrict__ yh) {
  const int b = blockIdx.z, h = blockIdx.y, qt = blockIdx.x;
  const int tid = threadIdx.x, wid = tid >> 5, lane = tid & 31;
  const int lm = lane & 15, lg = lane >> 4;
  const int i_base = qt * 16;
  const int col_base = wid * 128;

  __shared__ half_t sP[16][kT];     // normalized probs (f16) for P*V
  __shared__ float  sRed[8][16];    // per-wave row partials
  __shared__ float  sRowMax[16];
  __shared__ float  sRowSum[16];
  __shared__ float  sO[16][kHS];    // P*V accumulator

  for (int i = tid; i < 16 * kHS; i += 256) ((float*)sO)[i] = 0.0f;

  const int qrow = 3 * kC;  // 3072
  const half_t* qbase = qkv + (size_t)b * kT * qrow + h * kHS;  // + t*3072
  const half_t* kbase = qbase + kC;
  const bool active = (col_base <= i_base + 15);  // wave-uniform
  const float slope = exp2f(-0.5f * (float)h);    // 2^(-(8/H)*h), H=16
  const float scale = 0.125f;                     // 1/sqrt(64)

  v8f sc[8];
  float wmax[8];
#pragma unroll
  for (int r = 0; r < 8; ++r) wmax[r] = -__builtin_inff();

  if (active) {
    // S = Q Kt : A = Q(16x64), B[k=d][n=j] = K[j,d] -> row-major K rows.
    v16h qa0 = load_frag(qbase, i_base, qrow, 0, lane);
    v16h qa1 = load_frag(qbase, i_base, qrow, 32, lane);
#pragma unroll
    for (int nt = 0; nt < 8; ++nt) {
      const int j0 = col_base + nt * 16;
      v8f c = (v8f){};
      c = wmma_f16(qa0, load_frag(kbase, j0, qrow, 0, lane), c);
      c = wmma_f16(qa1, load_frag(kbase, j0, qrow, 32, lane), c);
      sc[nt] = c;
    }
    // scale + ALiBi + causal mask, per-element
#pragma unroll
    for (int nt = 0; nt < 8; ++nt) {
      const int j = col_base + nt * 16 + lm;
#pragma unroll
      for (int r = 0; r < 8; ++r) {
        const int i = i_base + r + 8 * lg;
        float s = sc[nt][r] * scale + slope * (float)(j - i);
        s = (j <= i) ? s : -__builtin_inff();
        sc[nt][r] = s;
        wmax[r] = fmaxf(wmax[r], s);
      }
    }
  }

  // row max: 16 lanes share a row -> xor-shuffle width 16, then cross-wave LDS
#pragma unroll
  for (int r = 0; r < 8; ++r)
#pragma unroll
    for (int m = 1; m < 16; m <<= 1)
      wmax[r] = fmaxf(wmax[r], __shfl_xor(wmax[r], m, 16));
  if (lm == 0)
#pragma unroll
    for (int r = 0; r < 8; ++r) sRed[wid][r + 8 * lg] = wmax[r];
  __syncthreads();
  if (tid < 16) {
    float m = -__builtin_inff();
#pragma unroll
    for (int w = 0; w < 8; ++w) m = fmaxf(m, sRed[w][tid]);
    sRowMax[tid] = m;
  }
  __syncthreads();

  float rmax[8], wsum[8];
#pragma unroll
  for (int r = 0; r < 8; ++r) { rmax[r] = sRowMax[r + 8 * lg]; wsum[r] = 0.0f; }

  if (active) {
#pragma unroll
    for (int nt = 0; nt < 8; ++nt)
#pragma unroll
      for (int r = 0; r < 8; ++r) {
        const float e = __expf(sc[nt][r] - rmax[r]);  // exp(-inf)=0 for masked
        sc[nt][r] = e;
        wsum[r] += e;
      }
  }
#pragma unroll
  for (int r = 0; r < 8; ++r)
#pragma unroll
    for (int m = 1; m < 16; m <<= 1) wsum[r] += __shfl_xor(wsum[r], m, 16);
  __syncthreads();  // sRed reuse
  if (lm == 0)
#pragma unroll
    for (int r = 0; r < 8; ++r) sRed[wid][r + 8 * lg] = wsum[r];
  __syncthreads();
  if (tid < 16) {
    float s = 0.0f;
#pragma unroll
    for (int w = 0; w < 8; ++w) s += sRed[w][tid];
    sRowSum[tid] = s;
  }
  __syncthreads();

  float rinv[8];
#pragma unroll
  for (int r = 0; r < 8; ++r) rinv[r] = 1.0f / sRowSum[r + 8 * lg];

  // write attn (f32, streamed once) + stash f16 probs in LDS
  const size_t attn_base = (((size_t)b * kH + h) * kT + i_base) * (size_t)kT;
  if (active) {
#pragma unroll
    for (int nt = 0; nt < 8; ++nt) {
      const int j = col_base + nt * 16 + lm;
#pragma unroll
      for (int r = 0; r < 8; ++r) {
        const int irow = r + 8 * lg;
        const float p = sc[nt][r] * rinv[r];
        attn[attn_base + (size_t)irow * kT + j] = p;
        sP[irow][j] = (half_t)p;
      }
    }
  } else {
#pragma unroll
    for (int nt = 0; nt < 8; ++nt) {
      const int j = col_base + nt * 16 + lm;
#pragma unroll
      for (int r = 0; r < 8; ++r)
        attn[attn_base + (size_t)(r + 8 * lg) * kT + j] = 0.0f;
    }
  }
  __syncthreads();

  // P*V: wave w reduces over its own K-chunk (its P columns); masked waves'
  // chunks are all-zero -> skipped. B[k=j][n=d] = V[j,d] = Vt[d][j] rows.
  if (active) {
    const half_t* vtb = vt + ((size_t)b * kH + h) * (size_t)kHS * kT;  // [64][T]
#pragma unroll
    for (int dt = 0; dt < 4; ++dt) {
      v8f o = (v8f){};
#pragma unroll
      for (int ks = 0; ks < 4; ++ks) {
        const int kb = col_base + ks * 32;
        v16h pa = load_frag(&sP[0][0], 0, kT, kb, lane);
        v16h vb = load_frag(vtb, dt * 16, kT, kb, lane);
        o = wmma_f16(pa, vb, o);
      }
#pragma unroll
      for (int r = 0; r < 8; ++r)
        atomicAdd(&sO[r + 8 * lg][dt * 16 + lm], o[r]);
    }
  }
  __syncthreads();

  // write Y (f16) for the output projection
  for (int i = tid; i < 16 * kHS; i += 256) {
    const int irow = i >> 6, d = i & 63;
    yh[((size_t)b * kT + i_base + irow) * kC + h * kHS + d] =
        (half_t)((const float*)sO)[i];
  }
}

// ---------------------------------------------------------------------------
// launch
// ---------------------------------------------------------------------------
extern "C" void kernel_launch(void* const* d_in, const int* in_sizes, int n_in,
                              void* d_out, int out_size, void* d_ws, size_t ws_size,
                              hipStream_t stream) {
  (void)in_sizes; (void)n_in; (void)out_size; (void)ws_size;
  const float* x     = (const float*)d_in[0];
  // d_in[1] = causal mask (int32) — mask is known causal, applied analytically
  const float* Wqkv  = (const float*)d_in[2];
  const float* bqkv  = (const float*)d_in[3];
  const float* Wproj = (const float*)d_in[4];
  const float* bproj = (const float*)d_in[5];

  float* out  = (float*)d_out;                        // [B,T,C]
  float* attn = out + (size_t)kB * kT * kC;           // [B,H,T,T]

  // workspace (~59 MB)
  half_t* xh   = (half_t*)d_ws;                       // B*T*C
  half_t* wqh  = xh   + (size_t)kB * kT * kC;         // 3C*C
  half_t* wph  = wqh  + (size_t)3 * kC * kC;          // C*C
  half_t* qkvh = wph  + (size_t)kC * kC;              // B*T*3C
  half_t* vth  = qkvh + (size_t)kB * kT * 3 * kC;     // B*H*64*T
  half_t* yh   = vth  + (size_t)kB * kH * kHS * kT;   // B*T*C

  const int nx = kB * kT * kC, nwq = 3 * kC * kC, nwp = kC * kC;
  mha_cvt_f32_f16<<<nx / 256, 256, 0, stream>>>(x, xh, nx);
  mha_cvt_f32_f16<<<nwq / 256, 256, 0, stream>>>(Wqkv, wqh, nwq);
  mha_cvt_f32_f16<<<nwp / 256, 256, 0, stream>>>(Wproj, wph, nwp);

  // qkv = x @ Wqkv^T + bqkv  (M=4096, N=3072, K=1024), f16 out
  mha_gemm_wmma<true><<<dim3(3 * kC / 128, kB * kT / 128), 256, 0, stream>>>(
      xh, wqh, bqkv, qkvh, kB * kT, 3 * kC, kC);

  mha_vtrans<<<(kB * kH * kHS * kT) / 256, 256, 0, stream>>>(qkvh, vth);

  mha_attn_kernel<<<dim3(kT / 16, kH, kB), 256, 0, stream>>>(qkvh, vth, attn, yh);

  // out = Y @ Wproj^T + bproj  (M=4096, N=1024, K=1024), f32 out
  mha_gemm_wmma<false><<<dim3(kC / 128, kB * kT / 128), 256, 0, stream>>>(
      yh, wph, bproj, out, kB * kT, kC, kC);
}
